// ModuleLayer_64338610094346
// MI455X (gfx1250) — matrix operand order repaired and checked by reading, not compiled, not run on hardware
//
#include <hip/hip_runtime.h>
#include <math.h>

typedef __bf16 bf16;
typedef __attribute__((ext_vector_type(16))) __bf16 v16bf;
typedef __attribute__((ext_vector_type(8)))  float  v8f;

#define B_   4
#define TY_  2048
#define TX_  2048
#define D_   1024
#define H_   16
#define HD_  64

// ---- CDNA5 async global->LDS staging (guarded: falls back to sync copy) ----
#if defined(__has_builtin)
#if __has_builtin(__builtin_amdgcn_global_load_async_to_lds_b128)
#define USE_ASYNC_LDS 1
#endif
#endif
#ifndef USE_ASYNC_LDS
#define USE_ASYNC_LDS 0
#endif

// Copies 16 bytes global -> LDS. With async path, completion is tracked by
// ASYNCcnt; callers must async_join() + __syncthreads() before reading LDS.
__device__ inline void async_cp16(const void* gsrc, void* ldst) {
#if USE_ASYNC_LDS
  // VGLOBAL async-to-LDS form: vdst = LDS byte address, vaddr = 64-bit global.
  // Generic pointers to __shared__ are aperture|offset; low 32 bits = LDS addr.
  const unsigned lds_off = (unsigned)(unsigned long long)ldst;
  asm volatile("global_load_async_to_lds_b128 %0, %1, off"
               :
               : "v"(lds_off), "v"((unsigned long long)gsrc)
               : "memory");
#else
  *(ulonglong2*)ldst = *(const ulonglong2*)gsrc;
#endif
}
__device__ inline void async_join() {
#if USE_ASYNC_LDS
#if __has_builtin(__builtin_amdgcn_s_wait_asynccnt)
  __builtin_amdgcn_s_wait_asynccnt(0);
#else
  asm volatile("s_wait_asynccnt 0" ::: "memory");
#endif
#endif
}

union BF16x16 { v16bf v; ulonglong2 q[2]; };

__device__ inline v8f vzero8() { v8f z = {0.f,0.f,0.f,0.f,0.f,0.f,0.f,0.f}; return z; }

// A-fragment (16x32 bf16): lane m=l16 holds 16 contiguous K starting at half*16
__device__ inline v16bf ld_afrag(const bf16* p) {
  BF16x16 u;
  u.q[0] = *(const ulonglong2*)(p);
  u.q[1] = *(const ulonglong2*)(p + 8);
  return u.v;
}
// B-fragment (32x16 bf16): lane n=l16 holds 16 contiguous K starting at half*16
__device__ inline v16bf ld_bfrag(const bf16* p) {
  BF16x16 u;
  u.q[0] = *(const ulonglong2*)(p);
  u.q[1] = *(const ulonglong2*)(p + 8);
  return u.v;
}

// ---------------------------------------------------------------------------
// Weight convert + transpose: Wt[n][k] = (bf16) W[k][n].  block (32,8)
// ---------------------------------------------------------------------------
__global__ __launch_bounds__(256) void k_wcvt_t(const float* __restrict__ W,
                                                bf16* __restrict__ Wt,
                                                int K, int N) {
  __shared__ float tile[32][33];
  const int k0 = blockIdx.y * 32;
  const int n0 = blockIdx.x * 32;
  const int tx = threadIdx.x, ty = threadIdx.y;
  for (int r = ty; r < 32; r += 8)
    tile[r][tx] = W[(size_t)(k0 + r) * N + n0 + tx];
  __syncthreads();
  for (int r = ty; r < 32; r += 8)
    Wt[(size_t)(n0 + r) * K + k0 + tx] = (bf16)tile[tx][r];
}

// ---------------------------------------------------------------------------
// GEMM: C[M,N] = A[M,K] @ Wt[N,K](bf16, pre-transposed) + bias (+GELU)
// AT: float (convert-stage) or bf16 (async byte-copy stage). OT: float or bf16.
// 128x128 block, k-tile 64 (2 WMMA k-steps per stage), 8 waves each 32x64.
// ---------------------------------------------------------------------------
template <typename AT, typename OT, int EPI>
__global__ __launch_bounds__(256) void k_gemm(const AT* __restrict__ A,
                                              const bf16* __restrict__ Wt,
                                              const float* __restrict__ bias,
                                              OT* __restrict__ C,
                                              int M, int N, int K) {
  __shared__ alignas(16) bf16 As[128 * 72];
  __shared__ alignas(16) bf16 Bs[128 * 72];
  const int tid  = threadIdx.x;
  const int m0   = blockIdx.y * 128;
  const int n0   = blockIdx.x * 128;
  const int wave = tid >> 5;
  const int lane = tid & 31;
  const int l16  = lane & 15;
  const int half = lane >> 4;
  const int wm   = wave >> 1;   // 0..3 -> 32-row strip
  const int wn   = wave & 1;    // 0..1 -> 64-col strip

  v8f acc[2][4];
#pragma unroll
  for (int i = 0; i < 2; ++i)
#pragma unroll
    for (int j = 0; j < 4; ++j) acc[i][j] = vzero8();

  for (int ko = 0; ko < K; ko += 64) {
    __syncthreads();
    // B tile: 128 rows x 64 bf16 = 1024 x 16B chunks, 4 per thread (async)
#pragma unroll
    for (int i = 0; i < 4; ++i) {
      const int lin = i * 256 + tid;
      const int r = lin >> 3;
      const int c = (lin & 7) * 8;
      async_cp16(Wt + (size_t)(n0 + r) * K + ko + c, &Bs[r * 72 + c]);
    }
    if constexpr (sizeof(AT) == 2) {
      // A already bf16: pure async byte copy
#pragma unroll
      for (int i = 0; i < 4; ++i) {
        const int lin = i * 256 + tid;
        const int r = lin >> 3;
        const int c = (lin & 7) * 8;
        async_cp16(A + (size_t)(m0 + r) * K + ko + c, &As[r * 72 + c]);
      }
    } else {
      // A f32: load, convert to bf16 into LDS. 128x64 f32 = 2048 float4.
#pragma unroll
      for (int i = 0; i < 8; ++i) {
        const int lin = i * 256 + tid;
        const int r = lin >> 4;
        const int c = (lin & 15) * 4;
        const float4 va = *(const float4*)((const float*)A + (size_t)(m0 + r) * K + ko + c);
        bf16* da = &As[r * 72 + c];
        da[0] = (bf16)va.x; da[1] = (bf16)va.y;
        da[2] = (bf16)va.z; da[3] = (bf16)va.w;
      }
      if (ko + 64 < K)
        __builtin_prefetch((const float*)A + (size_t)(m0 + (tid >> 4)) * K + ko + 64, 0, 1);
    }
    async_join();
    __syncthreads();

#pragma unroll
    for (int kc = 0; kc < 2; ++kc) {
      v16bf af[2];
#pragma unroll
      for (int mt = 0; mt < 2; ++mt)
        af[mt] = ld_afrag(&As[(wm * 32 + mt * 16 + l16) * 72 + kc * 32 + half * 16]);
#pragma unroll
      for (int nt = 0; nt < 4; ++nt) {
        const v16bf bfg = ld_bfrag(&Bs[(wn * 64 + nt * 16 + l16) * 72 + kc * 32 + half * 16]);
#pragma unroll
        for (int mt = 0; mt < 2; ++mt)
          acc[mt][nt] = __builtin_amdgcn_wmma_f32_16x16x32_bf16(
              false, af[mt], false, bfg, (short)0, acc[mt][nt], false, false);
      }
    }
  }

#pragma unroll
  for (int mt = 0; mt < 2; ++mt) {
#pragma unroll
    for (int nt = 0; nt < 4; ++nt) {
      const int col = n0 + wn * 64 + nt * 16 + l16;
      const float bv = bias ? bias[col] : 0.f;
#pragma unroll
      for (int v = 0; v < 8; ++v) {
        const int row = m0 + wm * 32 + mt * 16 + v + 8 * half;
        float x = acc[mt][nt][v] + bv;
        if (EPI == 1) {
          const float u = 0.7978845608028654f * (x + 0.044715f * x * x * x);
          x = 0.5f * x * (1.0f + tanhf(u));
        }
        C[(size_t)row * N + col] = (OT)x;
      }
    }
  }
}

// ---------------------------------------------------------------------------
// Flash cross-attention on bf16 q/k/v streams. Block = 64 queries x head x batch.
// 128 threads = 4 waves, each wave 16 query rows. Score scale folded into S.
// ---------------------------------------------------------------------------
__global__ __launch_bounds__(128) void k_flash(const bf16* __restrict__ qkv,
                                               const bf16* __restrict__ kvx,
                                               const float* __restrict__ amask,
                                               bf16* __restrict__ ctx) {
  __shared__ alignas(16) bf16 Qs[64 * 72];   // [q][hd]
  __shared__ alignas(16) bf16 Ks[64 * 72];   // [key][hd]
  __shared__ alignas(16) bf16 Vts[64 * 72];  // [hd][key]
  __shared__ alignas(16) bf16 Ps[64 * 72];   // [q][key]
  __shared__ float Ms[64];

  const int tid  = threadIdx.x;
  const int t0   = blockIdx.x * 64;
  const int h    = blockIdx.y;
  const int b    = blockIdx.z;
  const int wave = tid >> 5, lane = tid & 31, l16 = lane & 15, half = lane >> 4;

  // stage Q tile: 64x64 bf16 = 512 x 16B chunks, 4 per thread (async)
#pragma unroll
  for (int i = 0; i < 4; ++i) {
    const int lin = i * 128 + tid;
    const int r = lin >> 3;
    const int c = (lin & 7) * 8;
    async_cp16(qkv + (size_t)(b * TY_ + t0 + r) * (3 * D_) + h * HD_ + c, &Qs[r * 72 + c]);
  }

  float m8[8], l8[8];
  v8f O[4];
#pragma unroll
  for (int i = 0; i < 8; ++i) { m8[i] = -3.0e38f; l8[i] = 0.f; }
#pragma unroll
  for (int i = 0; i < 4; ++i) O[i] = vzero8();

  for (int s0 = 0; s0 < TX_; s0 += 64) {
    __syncthreads();
    // K tile async copy
#pragma unroll
    for (int i = 0; i < 4; ++i) {
      const int lin = i * 128 + tid;
      const int r = lin >> 3;
      const int c = (lin & 7) * 8;
      async_cp16(kvx + (size_t)(b * TX_ + s0 + r) * (3 * D_) + D_ + h * HD_ + c, &Ks[r * 72 + c]);
    }
    // V tile transposed into LDS (manual: 4 bf16 per thread per iter)
#pragma unroll
    for (int i = 0; i < 8; ++i) {
      const int lin = i * 128 + tid;
      const int r = lin >> 4;
      const int c = (lin & 15) * 4;
      union { unsigned long long u; bf16 e[4]; } vv;
      vv.u = *(const unsigned long long*)(kvx + (size_t)(b * TX_ + s0 + r) * (3 * D_) + 2 * D_ + h * HD_ + c);
      Vts[(c + 0) * 72 + r] = vv.e[0];
      Vts[(c + 1) * 72 + r] = vv.e[1];
      Vts[(c + 2) * 72 + r] = vv.e[2];
      Vts[(c + 3) * 72 + r] = vv.e[3];
    }
    if (tid < 64) Ms[tid] = amask[(size_t)b * TX_ + s0 + tid];
    async_join();
    __syncthreads();

    // S = (Q @ K^T) * 1/8 + mask
    v8f S[4];
#pragma unroll
    for (int nt = 0; nt < 4; ++nt) S[nt] = vzero8();
#pragma unroll
    for (int kc = 0; kc < 2; ++kc) {
      const v16bf aq = ld_afrag(&Qs[(wave * 16 + l16) * 72 + kc * 32 + half * 16]);
#pragma unroll
      for (int nt = 0; nt < 4; ++nt) {
        const v16bf bk = ld_bfrag(&Ks[(nt * 16 + l16) * 72 + kc * 32 + half * 16]);
        S[nt] = __builtin_amdgcn_wmma_f32_16x16x32_bf16(
            false, aq, false, bk, (short)0, S[nt], false, false);
      }
    }

    float mx[8];
#pragma unroll
    for (int v = 0; v < 8; ++v) mx[v] = -3.0e38f;
#pragma unroll
    for (int nt = 0; nt < 4; ++nt) {
      const float mk = Ms[nt * 16 + l16];
#pragma unroll
      for (int v = 0; v < 8; ++v) {
        S[nt][v] = S[nt][v] * 0.125f + mk;
        mx[v] = fmaxf(mx[v], S[nt][v]);
      }
    }
#pragma unroll
    for (int off = 8; off >= 1; off >>= 1)
#pragma unroll
      for (int v = 0; v < 8; ++v)
        mx[v] = fmaxf(mx[v], __shfl_xor(mx[v], off, 32));

    float alpha[8], rs[8];
#pragma unroll
    for (int v = 0; v < 8; ++v) {
      const float mnew = fmaxf(m8[v], mx[v]);
      alpha[v] = __expf(m8[v] - mnew);
      m8[v] = mnew;
      rs[v] = 0.f;
    }
    // P = exp(S - m); transpose C-layout -> A-layout via per-wave LDS region
#pragma unroll
    for (int nt = 0; nt < 4; ++nt)
#pragma unroll
      for (int v = 0; v < 8; ++v) {
        const float p = __expf(S[nt][v] - m8[v]);
        rs[v] += p;
        Ps[(wave * 16 + v + 8 * half) * 72 + nt * 16 + l16] = (bf16)p;
      }
#pragma unroll
    for (int off = 8; off >= 1; off >>= 1)
#pragma unroll
      for (int v = 0; v < 8; ++v)
        rs[v] += __shfl_xor(rs[v], off, 32);
#pragma unroll
    for (int v = 0; v < 8; ++v) l8[v] = l8[v] * alpha[v] + rs[v];
#pragma unroll
    for (int ht = 0; ht < 4; ++ht)
#pragma unroll
      for (int v = 0; v < 8; ++v) O[ht][v] *= alpha[v];

    // O += P @ V
#pragma unroll
    for (int kc = 0; kc < 2; ++kc) {
      const v16bf ap = ld_afrag(&Ps[(wave * 16 + l16) * 72 + kc * 32 + half * 16]);
#pragma unroll
      for (int ht = 0; ht < 4; ++ht) {
        const v16bf bv = ld_bfrag(&Vts[(ht * 16 + l16) * 72 + kc * 32 + half * 16]);
        O[ht] = __builtin_amdgcn_wmma_f32_16x16x32_bf16(
            false, ap, false, bv, (short)0, O[ht], false, false);
      }
    }
  }

#pragma unroll
  for (int v = 0; v < 8; ++v) {
    const float inv = 1.0f / l8[v];
    const int row = t0 + wave * 16 + v + 8 * half;
#pragma unroll
    for (int ht = 0; ht < 4; ++ht)
      ctx[(size_t)(b * TY_ + row) * D_ + h * HD_ + ht * 16 + l16] = (bf16)(O[ht][v] * inv);
  }
}

// ---------------------------------------------------------------------------
// out = LN(a + r) * g + b   (one block per row, D=1024, 256 threads)
// ---------------------------------------------------------------------------
__global__ __launch_bounds__(256) void k_ln_add(const float* __restrict__ a,
                                                const float* __restrict__ r,
                                                const float* __restrict__ g,
                                                const float* __restrict__ bta,
                                                float* __restrict__ out) {
  const int row = blockIdx.x;
  const size_t base = (size_t)row * D_;
  const int tid = threadIdx.x;
  __shared__ float red[256];
  float v[4]; float s = 0.f;
#pragma unroll
  for (int i = 0; i < 4; ++i) {
    const int c = tid + i * 256;
    v[i] = a[base + c] + r[base + c];
    s += v[i];
  }
  red[tid] = s; __syncthreads();
  for (int st = 128; st > 0; st >>= 1) { if (tid < st) red[tid] += red[tid + st]; __syncthreads(); }
  const float mu = red[0] * (1.f / 1024.f);
  __syncthreads();
  float s2 = 0.f;
#pragma unroll
  for (int i = 0; i < 4; ++i) { const float d = v[i] - mu; s2 += d * d; }
  red[tid] = s2; __syncthreads();
  for (int st = 128; st > 0; st >>= 1) { if (tid < st) red[tid] += red[tid + st]; __syncthreads(); }
  const float rstd = rsqrtf(red[0] * (1.f / 1024.f) + 1e-5f);
#pragma unroll
  for (int i = 0; i < 4; ++i) {
    const int c = tid + i * 256;
    out[base + c] = (v[i] - mu) * rstd * g[c] + bta[c];
  }
}

// out = LN(y2 + gate*highway) — gate/highway broadcast over rows of each batch
__global__ __launch_bounds__(256) void k_ln_final(const float* __restrict__ y2,
                                                  const float* __restrict__ hwy,
                                                  const float* __restrict__ gate,
                                                  const float* __restrict__ g,
                                                  const float* __restrict__ bta,
                                                  float* __restrict__ out) {
  const int row = blockIdx.x;
  const int b = row >> 11;  // / TY_
  const size_t base = (size_t)row * D_;
  const int tid = threadIdx.x;
  __shared__ float red[256];
  float v[4]; float s = 0.f;
#pragma unroll
  for (int i = 0; i < 4; ++i) {
    const int c = tid + i * 256;
    v[i] = y2[base + c] + gate[b * D_ + c] * hwy[b * D_ + c];
    s += v[i];
  }
  red[tid] = s; __syncthreads();
  for (int st = 128; st > 0; st >>= 1) { if (tid < st) red[tid] += red[tid + st]; __syncthreads(); }
  const float mu = red[0] * (1.f / 1024.f);
  __syncthreads();
  float s2 = 0.f;
#pragma unroll
  for (int i = 0; i < 4; ++i) { const float d = v[i] - mu; s2 += d * d; }
  red[tid] = s2; __syncthreads();
  for (int st = 128; st > 0; st >>= 1) { if (tid < st) red[tid] += red[tid + st]; __syncthreads(); }
  const float rstd = rsqrtf(red[0] * (1.f / 1024.f) + 1e-5f);
#pragma unroll
  for (int i = 0; i < 4; ++i) {
    const int c = tid + i * 256;
    out[base + c] = (v[i] - mu) * rstd * g[c] + bta[c];
  }
}

// pooled[b][d] = mean_t x_drv[b][t][d]
__global__ __launch_bounds__(256) void k_pool(const float* __restrict__ x,
                                              float* __restrict__ pooled) {
  const int d = blockIdx.x * 256 + threadIdx.x;
  const int b = blockIdx.y;
  float s = 0.f;
  for (int t = 0; t < TX_; ++t)
    s += x[(size_t)(b * TX_ + t) * D_ + d];
  pooled[b * D_ + d] = s * (1.f / (float)TX_);
}

// highway = pooled@hw_w + hw_b ; gate = sigmoid(pooled@gate_w + gate_b)
__global__ __launch_bounds__(256) void k_hwygate(const float* __restrict__ pooled,
                                                 const float* __restrict__ hw_w,
                                                 const float* __restrict__ hw_b,
                                                 const float* __restrict__ gw,
                                                 const float* __restrict__ gb,
                                                 float* __restrict__ hwy,
                                                 float* __restrict__ gate) {
  const int n = blockIdx.x * 256 + threadIdx.x;
  const int b = blockIdx.y;
  float a1 = hw_b[n], a2 = gb[n];
  for (int k = 0; k < D_; ++k) {
    const float p = pooled[b * D_ + k];
    a1 += p * hw_w[(size_t)k * D_ + n];
    a2 += p * gw[(size_t)k * D_ + n];
  }
  hwy[b * D_ + n] = a1;
  gate[b * D_ + n] = 1.f / (1.f + __expf(-a2));
}

// ---------------------------------------------------------------------------
extern "C" void kernel_launch(void* const* d_in, const int* in_sizes, int n_in,
                              void* d_out, int out_size, void* d_ws, size_t ws_size,
                              hipStream_t stream) {
  (void)in_sizes; (void)n_in; (void)out_size; (void)ws_size;
  const float* y        = (const float*)d_in[0];
  const float* x_det    = (const float*)d_in[1];
  const float* amask    = (const float*)d_in[2];
  const float* kv_w     = (const float*)d_in[3];
  const float* c_attn_w = (const float*)d_in[4];
  const float* c_attn_b = (const float*)d_in[5];
  const float* c_proj_w = (const float*)d_in[6];
  const float* c_proj_b = (const float*)d_in[7];
  const float* fc_w     = (const float*)d_in[8];
  const float* fc_b     = (const float*)d_in[9];
  const float* proj_w   = (const float*)d_in[10];
  const float* proj_b   = (const float*)d_in[11];
  const float* ln1_g    = (const float*)d_in[12];
  const float* ln1_b    = (const float*)d_in[13];
  const float* ln2_g    = (const float*)d_in[14];
  const float* ln2_b    = (const float*)d_in[15];
  const float* hw_w     = (const float*)d_in[16];
  const float* hw_b     = (const float*)d_in[17];
  const float* gate_w   = (const float*)d_in[18];
  const float* gate_b   = (const float*)d_in[19];
  const float* lnb_g    = (const float*)d_in[20];
  const float* lnb_b    = (const float*)d_in[21];
  float* out = (float*)d_out;

  char* ws = (char*)d_ws;
  size_t o = 0;
  auto take = [&](size_t bytes) -> char* {
    char* p = ws + o;
    o += (bytes + 255) & ~(size_t)255;
    return p;
  };
  const int M = B_ * TY_;  // 8192
  bf16*  wt_kv    = (bf16*)take((size_t)D_ * D_ * 2);
  bf16*  wt_attn  = (bf16*)take((size_t)3 * D_ * D_ * 2);
  bf16*  wt_cproj = (bf16*)take((size_t)D_ * D_ * 2);
  bf16*  wt_fc    = (bf16*)take((size_t)4 * D_ * D_ * 2);
  bf16*  wt_proj  = (bf16*)take((size_t)4 * D_ * D_ * 2);
  float* x_drv    = (float*)take((size_t)M * D_ * 4);
  bf16*  qkv      = (bf16*)take((size_t)M * 3 * D_ * 2);
  bf16*  kvx      = (bf16*)take((size_t)M * 3 * D_ * 2);
  bf16*  ctx      = (bf16*)take((size_t)M * D_ * 2);
  float* tmp      = (float*)take((size_t)M * D_ * 4);   // attn_out, then t2
  float* y1       = (float*)take((size_t)M * D_ * 4);
  bf16*  hbuf     = (bf16*)take((size_t)M * 4 * D_ * 2);
  float* y2       = (float*)take((size_t)M * D_ * 4);
  float* pooled   = (float*)take((size_t)B_ * D_ * 4);
  float* hwy      = (float*)take((size_t)B_ * D_ * 4);
  float* gate     = (float*)take((size_t)B_ * D_ * 4);

  const dim3 tb32x8(32, 8);
  // 1) weight convert + transpose to bf16 [N][K]
  k_wcvt_t<<<dim3(D_ / 32,     D_ / 32),     tb32x8, 0, stream>>>(kv_w,     wt_kv,    D_,     D_);
  k_wcvt_t<<<dim3(3 * D_ / 32, D_ / 32),     tb32x8, 0, stream>>>(c_attn_w, wt_attn,  D_,     3 * D_);
  k_wcvt_t<<<dim3(D_ / 32,     D_ / 32),     tb32x8, 0, stream>>>(c_proj_w, wt_cproj, D_,     D_);
  k_wcvt_t<<<dim3(4 * D_ / 32, D_ / 32),     tb32x8, 0, stream>>>(fc_w,     wt_fc,    D_,     4 * D_);
  k_wcvt_t<<<dim3(D_ / 32,     4 * D_ / 32), tb32x8, 0, stream>>>(proj_w,   wt_proj,  4 * D_, D_);

  // 2) driver adapter: x_drv = x @ kv_w  (f32 out, feeds pooling exactly)
  k_gemm<float, float, 0><<<dim3(D_ / 128, M / 128), 256, 0, stream>>>(
      x_det, wt_kv, nullptr, x_drv, M, D_, D_);

  // 3) pooled driver features + gated bridge pieces
  k_pool<<<dim3(D_ / 256, B_), 256, 0, stream>>>(x_drv, pooled);
  k_hwygate<<<dim3(D_ / 256, B_), 256, 0, stream>>>(pooled, hw_w, hw_b, gate_w, gate_b, hwy, gate);

  // 4) fused QKV on both streams (bf16 outputs -> attention consumes directly)
  k_gemm<float, bf16, 0><<<dim3(3 * D_ / 128, M / 128), 256, 0, stream>>>(
      y, wt_attn, c_attn_b, qkv, M, 3 * D_, D_);
  k_gemm<float, bf16, 0><<<dim3(3 * D_ / 128, M / 128), 256, 0, stream>>>(
      x_drv, wt_attn, c_attn_b, kvx, M, 3 * D_, D_);

  // 5) flash cross-attention -> ctx (bf16)
  k_flash<<<dim3(TY_ / 64, H_, B_), 128, 0, stream>>>(qkv, kvx, amask, ctx);

  // 6) attn_out = ctx @ c_proj + b ; y1 = LN(y + attn_out)
  k_gemm<bf16, float, 0><<<dim3(D_ / 128, M / 128), 256, 0, stream>>>(
      ctx, wt_cproj, c_proj_b, tmp, M, D_, D_);
  k_ln_add<<<M, 256, 0, stream>>>(y, tmp, ln1_g, ln1_b, y1);

  // 7) MLP: h = gelu(y1 @ fc + b) (bf16);  t2 = h @ proj + b;  y2 = LN(y1 + t2)
  k_gemm<float, bf16, 1><<<dim3(4 * D_ / 128, M / 128), 256, 0, stream>>>(
      y1, wt_fc, fc_b, hbuf, M, 4 * D_, D_);
  k_gemm<bf16, float, 0><<<dim3(D_ / 128, M / 128), 256, 0, stream>>>(
      hbuf, wt_proj, proj_b, tmp, M, D_, 4 * D_);
  k_ln_add<<<M, 256, 0, stream>>>(y1, tmp, ln2_g, ln2_b, y2);

  // 8) out = LN(y2 + gate * highway)
  k_ln_final<<<M, 256, 0, stream>>>(y2, hwy, gate, lnb_g, lnb_b, out);
}
